// Prob_Quantizer_40243843563638
// MI455X (gfx1250) — matrix operand order repaired and checked by reading, not compiled, not run on hardware
//
#include <hip/hip_runtime.h>
#include <hip/hip_bf16.h>

typedef __attribute__((ext_vector_type(16))) __bf16 v16bf;
typedef __attribute__((ext_vector_type(8)))  float  v8f;

#define N_TOT   65536
#define DIM     512
#define NQ      8
#define NCODE   256
#define SUBD    64
#define SPLITS  64               // grid = NQ * SPLITS = 512 workgroups
#define ROWS_PER_STAGE 128       // 8 waves * 16 rows

union ABfrag {
    v16bf v;
    uint4 q[2];
};

__device__ __forceinline__ unsigned short f2bf(float f) {
    union { float f; unsigned u; } u;
    u.f = f;
    unsigned x = u.u;
    // round-to-nearest-even
    unsigned r = x + 0x7FFFu + ((x >> 16) & 1u);
    return (unsigned short)(r >> 16);
}

__global__ __launch_bounds__(256, 1)
void pq_argmin_kernel(const float* __restrict__ x,
                      const float* __restrict__ cb,
                      float* __restrict__ qx,
                      long long* __restrict__ ids) {
    __shared__ unsigned short s_cb[NCODE * SUBD];       // 32 KB bf16 codebook (one q)
    __shared__ float          s_c2[NCODE];              // 1 KB  f32 squared norms
    __shared__ unsigned short s_x[ROWS_PER_STAGE * SUBD]; // 16 KB bf16 x slice

    const int tid   = threadIdx.x;
    const int lane  = tid & 31;
    const int wave  = tid >> 5;
    const int q     = blockIdx.x & (NQ - 1);
    const int split = blockIdx.x >> 3;

    const int nl    = lane & 15;   // column-in-tile / row-in-A
    const int half  = lane >> 4;   // K-half selector per ISA layouts

    const float* __restrict__ cbq = cb + (size_t)q * NCODE * SUBD;

    // ---- stage codebook (one code per thread): f32 -> bf16 LDS + c2 ----
    {
        const int k = tid;                    // 256 threads, 256 codes
        const float* src = cbq + k * SUBD;
        unsigned short* dst = &s_cb[k * SUBD];
        float ss = 0.0f;
        #pragma unroll
        for (int j = 0; j < SUBD; j += 4) {
            float4 v = *(const float4*)(src + j);
            ss += v.x * v.x + v.y * v.y + v.z * v.z + v.w * v.w;
            dst[j + 0] = f2bf(v.x);
            dst[j + 1] = f2bf(v.y);
            dst[j + 2] = f2bf(v.z);
            dst[j + 3] = f2bf(v.w);
        }
        s_c2[k] = ss;
    }
    __syncthreads();

    const int rows_per_split = N_TOT / SPLITS;          // 1024
    const int row0 = split * rows_per_split;

    for (int stage = 0; stage < rows_per_split / ROWS_PER_STAGE; ++stage) {
        const int srow = row0 + stage * ROWS_PER_STAGE;

        // ---- stage x slice: 128 rows x 64 floats -> bf16 LDS ----
        {
            const int r = tid >> 1;            // row within stage (0..127)
            const int h = tid & 1;             // which 32-float half
            const float* src = x + (size_t)(srow + r) * DIM + q * SUBD + h * 32;
            unsigned short* dst = &s_x[r * SUBD + h * 32];
            #pragma unroll
            for (int j = 0; j < 32; j += 4) {
                float4 v = *(const float4*)(src + j);
                dst[j + 0] = f2bf(v.x);
                dst[j + 1] = f2bf(v.y);
                dst[j + 2] = f2bf(v.z);
                dst[j + 3] = f2bf(v.w);
            }
        }
        __syncthreads();

        // ---- A fragments for this wave's 16-row block (2 K-chunks) ----
        // ISA 16-bit A 16x32 layout: lane m = l%16; lanes 0-15 hold K = 0..7,16..23,
        // lanes 16-31 hold K = 8..15,24..31 (contiguous per 8-element group).
        ABfrag a0, a1;
        {
            const unsigned short* xr = &s_x[(wave * 16 + nl) * SUBD];
            a0.q[0] = *(const uint4*)(xr + 8 * half);            // K = 8h .. 8h+7
            a0.q[1] = *(const uint4*)(xr + 16 + 8 * half);       // K = 16+8h ..
            a1.q[0] = *(const uint4*)(xr + 32 + 8 * half);       // chunk 1 (K 32..63)
            a1.q[1] = *(const uint4*)(xr + 48 + 8 * half);
        }

        float minv[8];
        int   mini[8];
        #pragma unroll
        for (int r = 0; r < 8; ++r) { minv[r] = 3.4e38f; mini[r] = 0; }

        // ---- 16 column tiles of 16 codes, 2 WMMAs (K=64) each ----
        #pragma unroll 4
        for (int t = 0; t < 16; ++t) {
            // ISA 16-bit B KxN layout: lane n = l%16 holds column n,
            // K = 8*(l/16) + 0..7 and 16 + 8*(l/16) + 0..7 per chunk (matches A).
            ABfrag b0, b1;
            const unsigned short* cp = &s_cb[(t * 16 + nl) * SUBD + 8 * half];
            b0.q[0] = *(const uint4*)(cp);
            b0.q[1] = *(const uint4*)(cp + 16);
            b1.q[0] = *(const uint4*)(cp + 32);
            b1.q[1] = *(const uint4*)(cp + 48);

            v8f acc = {};
            acc = __builtin_amdgcn_wmma_f32_16x16x32_bf16(
                      false, a0.v, false, b0.v, (short)0, acc, false, false);
            acc = __builtin_amdgcn_wmma_f32_16x16x32_bf16(
                      false, a1.v, false, b1.v, (short)0, acc, false, false);

            const int   code = t * 16 + nl;
            const float c2v  = s_c2[code];
            #pragma unroll
            for (int r = 0; r < 8; ++r) {
                float sc = c2v - 2.0f * acc[r];
                if (sc < minv[r]) { minv[r] = sc; mini[r] = code; }
            }
        }

        // ---- min+argmin across the 16 lanes of each half (rows m=r+8*half) ----
        #pragma unroll
        for (int r = 0; r < 8; ++r) {
            float v = minv[r];
            int   i = mini[r];
            #pragma unroll
            for (int off = 8; off >= 1; off >>= 1) {
                float ov = __shfl_xor(v, off, 32);
                int   oi = __shfl_xor(i, off, 32);
                if (ov < v || (ov == v && oi < i)) { v = ov; i = oi; }
            }
            minv[r] = v;
            mini[r] = i;
        }

        const int rbase = srow + wave * 16;

        // ---- indices: lane nl==0 of each half holds rows (8*half + r) ----
        if (nl == 0) {
            const int base = rbase + half * 8;
            #pragma unroll
            for (int r = 0; r < 8; ++r)
                ids[(size_t)(base + r) * NQ + q] = (long long)mini[r];
        }

        // ---- quantized rows: copy winning f32 codebook rows (L2-resident) ----
        #pragma unroll
        for (int m = 0; m < 16; ++m) {
            const int k = __shfl(mini[m & 7], (m >> 3) * 16, 32);
            const float* crow = cbq + k * SUBD;
            float2 val = *(const float2*)(crow + lane * 2);
            *(float2*)(qx + (size_t)(rbase + m) * DIM + q * SUBD + lane * 2) = val;
        }

        __syncthreads();   // protect s_x before next stage overwrites it
    }
}

extern "C" void kernel_launch(void* const* d_in, const int* in_sizes, int n_in,
                              void* d_out, int out_size, void* d_ws, size_t ws_size,
                              hipStream_t stream) {
    (void)in_sizes; (void)n_in; (void)d_ws; (void)ws_size; (void)out_size;
    const float* x  = (const float*)d_in[0];
    const float* cb = (const float*)d_in[1];
    float*     qx  = (float*)d_out;
    long long* ids = (long long*)(qx + (size_t)N_TOT * DIM);

    dim3 grid(NQ * SPLITS);
    dim3 block(256);
    pq_argmin_kernel<<<grid, block, 0, stream>>>(x, cb, qx, ids);
}
